// LSTMStack_32452772888716
// MI455X (gfx1250) — matrix operand order repaired and checked by reading, not compiled, not run on hardware
//
#include <hip/hip_runtime.h>
#include <hip/hip_bf16.h>

// ---------------------------------------------------------------------------
// LSTM stack for MI455X (gfx1250, wave32, WMMA, async-to-LDS).
//   D=1024, L=4, T=512, B=32.
// Input GEMMs:      v_wmma_f32_16x16x32_bf16 (bf16 in, f32 acc)  - throughput
// Recurrent GEMMs:  v_wmma_f32_16x16x4_f32   (full fp32)         - precision
// Recurrent h tile staged to LDS with global_load_async_to_lds_b128
// (ASYNCcnt / s_wait_asynccnt), x_gates lines warmed with global_prefetch.
// Workspace needed: ~450 MB.
// ---------------------------------------------------------------------------

#define DD     1024
#define LAYERS 4
#define TT     512
#define BB     32
#define G4     (4 * DD)     // 4096 gate columns
#define ROWS   (TT * BB)    // 16384 rows for the batched input GEMM
#define LN_EPS 1e-5f

typedef __attribute__((ext_vector_type(16))) __bf16 bf16x16;
typedef __attribute__((ext_vector_type(8)))  __bf16 bf16x8;
typedef __attribute__((ext_vector_type(8)))  float  f32x8;
typedef __attribute__((ext_vector_type(2)))  float  f32x2;

// ---------------------------------------------------------------------------
// LayerNorm over the last dim (D=1024). One block (256 threads) per row.
// mode 0: write bf16 (feeds the WMMA input GEMM). mode 1: write f32 (final y).
// ---------------------------------------------------------------------------
__global__ __launch_bounds__(256) void ln_kernel(const float* __restrict__ in,
                                                 const float* __restrict__ gam,
                                                 const float* __restrict__ bet,
                                                 __bf16* __restrict__ out_bf,
                                                 float* __restrict__ out_f,
                                                 int mode) {
  const int row  = blockIdx.x;
  const int tid  = threadIdx.x;
  const int wave = tid >> 5, lane = tid & 31;
  const float* x = in + (size_t)row * DD;

  float v[4];
  float s = 0.f, ss = 0.f;
#pragma unroll
  for (int k = 0; k < 4; ++k) {
    v[k] = x[tid + k * 256];
    s += v[k];
    ss += v[k] * v[k];
  }
#pragma unroll
  for (int off = 16; off > 0; off >>= 1) {   // wave32 reduction
    s  += __shfl_down(s,  off, 32);
    ss += __shfl_down(ss, off, 32);
  }
  __shared__ float sred[8], ssred[8], stats[2];
  if (lane == 0) { sred[wave] = s; ssred[wave] = ss; }
  __syncthreads();
  if (tid == 0) {
    float S = 0.f, SS = 0.f;
    for (int i = 0; i < 8; ++i) { S += sred[i]; SS += ssred[i]; }
    float mean = S * (1.f / DD);
    float var  = SS * (1.f / DD) - mean * mean;
    stats[0] = mean;
    stats[1] = rsqrtf(var + LN_EPS);
  }
  __syncthreads();
  const float mean = stats[0], rstd = stats[1];
#pragma unroll
  for (int k = 0; k < 4; ++k) {
    int c = tid + k * 256;
    float y = (v[k] - mean) * rstd * gam[c] + bet[c];
    if (mode == 0) out_bf[(size_t)row * DD + c] = (__bf16)y;
    else           out_f [(size_t)row * DD + c] = y;
  }
}

// ---------------------------------------------------------------------------
__global__ void f32_to_bf16(const float* __restrict__ in,
                            __bf16* __restrict__ out, int n) {
  int i = blockIdx.x * blockDim.x + threadIdx.x;
  if (i < n) out[i] = (__bf16)in[i];
}

__global__ void init_hc(const float* __restrict__ h0,
                        const float* __restrict__ c0,
                        float* __restrict__ H, float* __restrict__ C) {
  int i = blockIdx.x * blockDim.x + threadIdx.x;
  if (i < BB * DD) { H[i] = h0[i]; C[i] = c0[i]; }
}

// ---------------------------------------------------------------------------
// Batched input GEMM: XG[16384,4096] = XN_bf16[16384,1024] @ W_ih^T + b_ih.
// Block = 8 waves -> 64(M) x 256(N) tile; each wave owns a 32x64 subtile
// (2 A-frags x 4 B-frags -> 8 WMMAs per K-step, K-loop of 32).
// bf16 A fragment layout (ISA 7.12.2): lane(m=l&15, hi=l>>4) holds
//   K = hi*8..hi*8+7 and K = 16+hi*8..+7   (two 16B chunks).
// bf16 B fragment: lane n=l&15 holds column n (= w_ih row), same K as A.
// ---------------------------------------------------------------------------
__global__ __launch_bounds__(256) void gemm_xgates(
    const __bf16* __restrict__ XN,   // [16384,1024]
    const __bf16* __restrict__ W,    // [4096,1024]
    const float*  __restrict__ bias, // [4096]
    float* __restrict__ XG) {        // [16384,4096]
  const int lane = threadIdx.x & 31;
  const int wave = threadIdx.x >> 5;
  const int wm = wave & 1, wn = wave >> 1;
  const int m0 = blockIdx.y * 64 + wm * 32;
  const int n0 = blockIdx.x * 256 + wn * 64;
  const int m = lane & 15, hi = lane >> 4;

  f32x8 acc[2][4];
#pragma unroll
  for (int mi = 0; mi < 2; ++mi)
#pragma unroll
    for (int ni = 0; ni < 4; ++ni)
#pragma unroll
      for (int j = 0; j < 8; ++j) acc[mi][ni][j] = 0.f;

  for (int kk = 0; kk < DD; kk += 32) {
    bf16x16 a[2], b[4];
#pragma unroll
    for (int mi = 0; mi < 2; ++mi) {
      const __bf16* p = XN + (size_t)(m0 + mi * 16 + m) * DD + kk + hi * 8;
      bf16x8 lo = *(const bf16x8*)p;
      bf16x8 hh = *(const bf16x8*)(p + 16);
#pragma unroll
      for (int i = 0; i < 8; ++i) { a[mi][i] = lo[i]; a[mi][i + 8] = hh[i]; }
    }
#pragma unroll
    for (int ni = 0; ni < 4; ++ni) {
      const __bf16* pb = W + (size_t)(n0 + ni * 16 + m) * DD + kk + hi * 8;
      bf16x8 blo = *(const bf16x8*)pb;
      bf16x8 bhi = *(const bf16x8*)(pb + 16);
#pragma unroll
      for (int i = 0; i < 8; ++i) { b[ni][i] = blo[i]; b[ni][i + 8] = bhi[i]; }
    }
#pragma unroll
    for (int mi = 0; mi < 2; ++mi)
#pragma unroll
      for (int ni = 0; ni < 4; ++ni)
        acc[mi][ni] = __builtin_amdgcn_wmma_f32_16x16x32_bf16(
            false, a[mi], false, b[ni], (short)0, acc[mi][ni], false, false);
  }

  // C/D layout: VGPR j, lanes 0-15 -> M=j ; lanes 16-31 -> M=j+8 ; N=lane&15.
#pragma unroll
  for (int mi = 0; mi < 2; ++mi)
#pragma unroll
    for (int ni = 0; ni < 4; ++ni) {
      int col = n0 + ni * 16 + m;
      float bv = bias[col];
#pragma unroll
      for (int j = 0; j < 8; ++j) {
        int row = m0 + mi * 16 + j + 8 * hi;
        XG[(size_t)row * G4 + col] = acc[mi][ni][j] + bv;
      }
    }
}

// ---------------------------------------------------------------------------
// One recurrent timestep, fused GEMM + LSTM cell.  Grid = 16 blocks; block b
// owns gate columns d0..d0+63 of ALL four gates -> can run the cell locally.
//
// Hprev (32x1024 fp32 = 128 KB) is staged into LDS with the CDNA5 async
// engine: global_load_async_to_lds_b128 (each instruction moves 32 lanes x
// 16 B), tracked by ASYNCcnt and drained with s_wait_asynccnt.  This removes
// the 4x redundant L2 reads of h (4 waves per batch-half share the same rows)
// and turns the per-K A-fragment fetches into ds_load_b64.  While the DMA is
// in flight we prefetch this block's x_gates[t] lines (global_prefetch_b8).
//
// 8 waves: wave = (mi batch-half, q gate).  Full fp32 WMMA 16x16x4:
//   A lane(m=l&15,hi=l>>4) holds K=2*hi,2*hi+1 ; B lane n holds col n, same K.
// Hprev/Hnext are ping-pong buffers; C is exclusive per (b,d) -> in-place.
// ---------------------------------------------------------------------------
__global__ __launch_bounds__(256) void lstm_step(
    const float* __restrict__ Hprev, // [32,1024]
    const float* __restrict__ Whh,   // [4096,1024] fp32
    const float* __restrict__ XGt,   // [32,4096]  (x_gates for this t)
    const float* __restrict__ bhh,   // [4096]
    float* __restrict__ Cst,         // [32,1024] in/out
    float* __restrict__ Hnext,       // [32,1024]
    float* __restrict__ Yt,          // [32,1024]  ys[t]
    float* __restrict__ outH,        // last-step h sink
    float* __restrict__ outC,        // last-step c sink
    int last) {
  __shared__ float hsh[BB * DD];     // staged h_{t-1}, 128 KB
  __shared__ float gl[32][256];      // 32 batch x (4 gates x 64 cols), 32 KB

  const int lane = threadIdx.x & 31;
  const int wave = threadIdx.x >> 5;
  const int mi = wave & 1;           // batch half: rows mi*16 .. mi*16+15
  const int q  = wave >> 1;          // gate 0..3
  const int d0 = blockIdx.x * 64;
  const int n = lane & 15, hi = lane >> 4;

  // ---- async DMA: Hprev -> LDS (8192 x b128 chunks over 256 threads) ----
  {
    // Flat LDS pointer truncated to 32 bits == LDS byte offset (ISA 10.2).
    unsigned lds_base = (unsigned)(uintptr_t)(&hsh[0]);
    for (int c = threadIdx.x; c < (BB * DD) / 4; c += 256) {
      unsigned loff = lds_base + c * 16u;
      unsigned long long ga = (unsigned long long)(uintptr_t)(Hprev + c * 4);
      asm volatile("global_load_async_to_lds_b128 %0, %1, off"
                   :: "v"(loff), "v"(ga) : "memory");
    }
  }

  // ---- overlap: warm the x_gates lines this block's cell phase will read --
  for (int e = threadIdx.x; e < 32 * 64; e += 256) {
    int b  = e >> 6, dd = e & 63;
    const float* xg = XGt + (size_t)b * G4 + d0 + dd;
    __builtin_prefetch(xg,          0, 1);
    __builtin_prefetch(xg +     DD, 0, 1);
    __builtin_prefetch(xg + 2 * DD, 0, 1);
    __builtin_prefetch(xg + 3 * DD, 0, 1);
  }

  // ---- drain DMA, make LDS visible to all waves ----
#if __has_builtin(__builtin_amdgcn_s_wait_asynccnt)
  __builtin_amdgcn_s_wait_asynccnt(0);
#else
  asm volatile("s_wait_asynccnt 0x0" ::: "memory");
#endif
  __syncthreads();

  f32x8 acc[4];
#pragma unroll
  for (int ni = 0; ni < 4; ++ni)
#pragma unroll
    for (int j = 0; j < 8; ++j) acc[ni][j] = 0.f;

  const float* Arow  = hsh + (size_t)(mi * 16 + n) * DD;  // A from LDS
  const float* Wbase = Whh + (size_t)(q * DD + d0) * DD;  // rows of W_hh

  for (int kk = 0; kk < DD; kk += 4) {
    f32x2 a = *(const f32x2*)(Arow + kk + 2 * hi);        // ds_load_b64
#pragma unroll
    for (int ni = 0; ni < 4; ++ni) {
      f32x2 b = *(const f32x2*)(Wbase + (size_t)(ni * 16 + n) * DD + kk + 2 * hi);
      acc[ni] = __builtin_amdgcn_wmma_f32_16x16x4_f32(
          false, a, false, b, (short)0, acc[ni], false, false);
    }
  }

#pragma unroll
  for (int ni = 0; ni < 4; ++ni)
#pragma unroll
    for (int j = 0; j < 8; ++j)
      gl[mi * 16 + j + 8 * hi][q * 64 + ni * 16 + n] = acc[ni][j];
  __syncthreads();

  // Fused LSTM cell over the 32x64 (b,d) elements this block owns.
  for (int e = threadIdx.x; e < 32 * 64; e += 256) {
    int b  = e >> 6, dd = e & 63;
    int d  = d0 + dd;
    const float* xg = XGt + (size_t)b * G4;
    float ir  = gl[b][      dd] + xg[           d] + bhh[           d];
    float fr  = gl[b][ 64 + dd] + xg[    DD   + d] + bhh[    DD   + d];
    float gr  = gl[b][128 + dd] + xg[2 * DD   + d] + bhh[2 * DD   + d];
    float orr = gl[b][192 + dd] + xg[3 * DD   + d] + bhh[3 * DD   + d];
    float ig = 1.f / (1.f + expf(-ir));
    float fg = 1.f / (1.f + expf(-fr));
    float gg = tanhf(gr);
    float og = 1.f / (1.f + expf(-orr));
    size_t idx = (size_t)b * DD + d;
    float c = fg * Cst[idx] + ig * gg;
    float h = og * tanhf(c);
    Cst[idx]   = c;
    Hnext[idx] = h;
    Yt[idx]    = h;
    if (last) { outH[idx] = h; outC[idx] = c; }
  }
}

// ---------------------------------------------------------------------------
extern "C" void kernel_launch(void* const* d_in, const int* in_sizes, int n_in,
                              void* d_out, int out_size, void* d_ws, size_t ws_size,
                              hipStream_t stream) {
  const float* x    = (const float*)d_in[0];   // [T,B,D]
  const float* h0   = (const float*)d_in[1];   // [L,B,D]
  const float* c0   = (const float*)d_in[2];   // [L,B,D]
  const float* w_ih = (const float*)d_in[3];   // [L,4D,D]
  const float* w_hh = (const float*)d_in[4];   // [L,4D,D]
  const float* b_ih = (const float*)d_in[5];   // [L,4D]
  const float* b_hh = (const float*)d_in[6];   // [L,4D]
  const float* ln_g = (const float*)d_in[7];   // [L,D]
  const float* ln_b = (const float*)d_in[8];   // [L,D]
  const float* st_g = (const float*)d_in[9];   // [D]
  const float* st_b = (const float*)d_in[10];  // [D]

  float* y_out = (float*)d_out;                      // [T,B,D]
  float* h_out = y_out + (size_t)TT * BB * DD;       // [L,B,D]
  float* c_out = h_out + (size_t)LAYERS * BB * DD;   // [L,B,D]

  // Carve workspace (256B aligned chunks).
  char*  ws  = (char*)d_ws;
  size_t off = 0;
  auto carve = [&](size_t bytes) -> char* {
    char* p = ws + off;
    off += (bytes + 255) & ~(size_t)255;
    return p;
  };
  __bf16* xn_bf  = (__bf16*)carve((size_t)ROWS * DD * 2);           //  32 MB
  __bf16* wih_bf = (__bf16*)carve((size_t)LAYERS * G4 * DD * 2);    //  32 MB
  float*  xg     = (float*) carve((size_t)ROWS * G4 * 4);           // 256 MB
  float*  ysA    = (float*) carve((size_t)ROWS * DD * 4);           //  64 MB
  float*  ysB    = (float*) carve((size_t)ROWS * DD * 4);           //  64 MB
  float*  Ha     = (float*) carve((size_t)BB * DD * 4);
  float*  Hb     = (float*) carve((size_t)BB * DD * 4);
  float*  Cs     = (float*) carve((size_t)BB * DD * 4);

  // Convert all w_ih weights to bf16 once.
  {
    int n = LAYERS * G4 * DD;
    f32_to_bf16<<<(n + 255) / 256, 256, 0, stream>>>(w_ih, wih_bf, n);
  }

  const float* lin = x;
  float* ybufs[2] = { ysA, ysB };
  for (int li = 0; li < LAYERS; ++li) {
    // 1) LayerNorm -> bf16 activations for the WMMA GEMM.
    ln_kernel<<<ROWS, 256, 0, stream>>>(lin, ln_g + li * DD, ln_b + li * DD,
                                        xn_bf, nullptr, 0);
    // 2) Batched input GEMM (+ b_ih).
    dim3 grid(G4 / 256, ROWS / 64);
    gemm_xgates<<<grid, 256, 0, stream>>>(xn_bf, wih_bf + (size_t)li * G4 * DD,
                                          b_ih + (size_t)li * G4, xg);
    // 3) Init recurrent state.
    init_hc<<<(BB * DD + 255) / 256, 256, 0, stream>>>(
        h0 + (size_t)li * BB * DD, c0 + (size_t)li * BB * DD, Ha, Cs);
    // 4) Sequential recurrence (ping-pong h buffers).
    float* ys = ybufs[li & 1];
    const float* whh_l = w_hh + (size_t)li * G4 * DD;
    for (int t = 0; t < TT; ++t) {
      const float* Hin  = (t & 1) ? Hb : Ha;
      float*       Hout = (t & 1) ? Ha : Hb;
      lstm_step<<<DD / 64, 256, 0, stream>>>(
          Hin, whh_l, xg + (size_t)t * BB * G4, b_hh + (size_t)li * G4,
          Cs, Hout, ys + (size_t)t * BB * DD,
          h_out + (size_t)li * BB * DD, c_out + (size_t)li * BB * DD,
          (t == TT - 1) ? 1 : 0);
    }
    lin = ys;
  }
  // 5) Final stack LayerNorm -> fp32 y.
  ln_kernel<<<ROWS, 256, 0, stream>>>(lin, st_g, st_b, nullptr, y_out, 1);
}
